// MMoE_72662256714401
// MI455X (gfx1250) — matrix operand order repaired and checked by reading, not compiled, not run on hardware
//
#include <hip/hip_runtime.h>
#include <hip/hip_bf16.h>

// MMoE fused forward for MI455X (gfx1250), wave32 + bf16 WMMA.
//
// out[t,b] = sum_e softmax_e(x.Wg[t,e]+bg)[b,t,e] * ( relu(x.We[e]^T+be[e]) . Wf[t] ) + bf[t]
//
// Expert GEMM (137 GFLOP) runs as v_wmma_f32_16x16x32_bf16 with the relu/Wf
// reduction fused into the epilogue so eo[B,E,H] (256 MB) never hits HBM.
// Inner loop keeps 4 independent WMMA accumulation chains (4 N-chunks per
// K-step) so A-fragment LDS loads are amortized 4x and WMMA latency is hidden.

typedef __attribute__((ext_vector_type(16))) unsigned short v16u;
typedef __attribute__((ext_vector_type(8)))  unsigned short v8u;
typedef __attribute__((ext_vector_type(4)))  unsigned short v4u;
typedef __attribute__((ext_vector_type(16))) __bf16        v16bf;
typedef __attribute__((ext_vector_type(8)))  float         v8f;

#define B_TOK   8192
#define I_DIM   1024
#define H_DIM   512
#define E_NUM   16
#define T_NUM   4
#define BM      128                       // token rows per workgroup
#define LDSROW  1032                      // padded LDS row stride in bf16 elems (2064 B -> 4-bank skew)
#define LDS_BYTES (BM * LDSROW * 2)       // 264192 B < 320 KB WGP LDS

// f32 -> bf16 round-to-nearest-even
__device__ __forceinline__ unsigned short f2bf(float f) {
  unsigned u = __builtin_bit_cast(unsigned, f);
  u += 0x7FFFu + ((u >> 16) & 1u);
  return (unsigned short)(u >> 16);
}

// Stage a 128 x 1024 f32 tile of x into LDS as bf16 (row-padded).
__device__ __forceinline__ void stage_x(unsigned short* xs, const float* __restrict__ x,
                                        int b0, int tid) {
  #pragma unroll 4
  for (int r = 0; r < BM; ++r) {                       // 256 threads * float4 == one 1024-col row
    const float4 f = *reinterpret_cast<const float4*>(x + (size_t)(b0 + r) * I_DIM + tid * 4);
    v4u h = { f2bf(f.x), f2bf(f.y), f2bf(f.z), f2bf(f.w) };
    *reinterpret_cast<v4u*>(xs + r * LDSROW + tid * 4) = h;
  }
}

// -------- Kernel 1: weight conversion f32 -> bf16 (We then Wg) --------
__global__ void mmoe_convert_kernel(const float* __restrict__ We,
                                    const float* __restrict__ Wg,
                                    unsigned short* __restrict__ WeB,
                                    unsigned short* __restrict__ WgB) {
  const size_t i4  = (size_t)blockIdx.x * blockDim.x + threadIdx.x;
  const size_t NW4 = (size_t)E_NUM * H_DIM * I_DIM / 4;   // 2,097,152
  const size_t NG4 = (size_t)T_NUM * E_NUM * I_DIM / 4;   //    16,384
  const float* src; unsigned short* dst; size_t idx;
  if (i4 < NW4)            { src = We; dst = WeB; idx = i4; }
  else if (i4 < NW4 + NG4) { src = Wg; dst = WgB; idx = i4 - NW4; }
  else return;
  const float4 f = *reinterpret_cast<const float4*>(src + idx * 4);
  v4u h = { f2bf(f.x), f2bf(f.y), f2bf(f.z), f2bf(f.w) };
  *reinterpret_cast<v4u*>(dst + idx * 4) = h;
}

// -------- Kernel 2: expert GEMM + fused relu/Wf reduction -> s[B,E,T] --------
__global__ __launch_bounds__(256, 1) void mmoe_expert_kernel(
    const float* __restrict__ x, const unsigned short* __restrict__ WeB,
    const float* __restrict__ be, const float* __restrict__ Wf,
    float* __restrict__ s_out) {
  extern __shared__ __align__(16) char smem[];
  unsigned short* xs = reinterpret_cast<unsigned short*>(smem);
  const int tid = threadIdx.x;
  const int b0  = blockIdx.x * BM;
  const int e   = blockIdx.y;

  stage_x(xs, x, b0, tid);
  __syncthreads();

  const int lane  = tid & 31;
  const int w     = tid >> 5;          // wave id -> M-subtile
  const int lhalf = lane >> 4;         // 0: lanes 0-15, 1: lanes 16-31
  const int lmod  = lane & 15;
  const unsigned short* xrow = xs + (w * 16 + lmod) * LDSROW;
  const int kA = lhalf * 16;           // A-frag K contiguous per lane half (matches B)

  float sacc[T_NUM][8];
  #pragma unroll
  for (int t = 0; t < T_NUM; ++t)
    #pragma unroll
    for (int r = 0; r < 8; ++r) sacc[t][r] = 0.f;

  // 8 groups x 4 N-chunks x 16 cols = H (512). 4 independent WMMA chains.
  for (int ng = 0; ng < H_DIM / 64; ++ng) {
    const int nbase = ng * 64;
    // Column (We row) for chain j is nbase + j*16 + lmod.
    const unsigned short* brow =
        WeB + ((size_t)e * H_DIM + nbase + lmod) * I_DIM + lhalf * 16;
    v8f c[4] = {};
    #pragma unroll 4
    for (int kc = 0; kc < I_DIM / 32; ++kc) {          // K = 1024, 32 per WMMA
      const int k0 = kc * 32 + kA;
      v8u alo = *reinterpret_cast<const v8u*>(xrow + k0);        // K 0-7 / 16-23
      v8u ahi = *reinterpret_cast<const v8u*>(xrow + k0 + 8);    // K 8-15 / 24-31
      v16u a  = __builtin_shufflevector(alo, ahi, 0,1,2,3,4,5,6,7,8,9,10,11,12,13,14,15);
      #pragma unroll
      for (int j = 0; j < 4; ++j) {
        v16u b = *reinterpret_cast<const v16u*>(brow + (size_t)j * 16 * I_DIM + kc * 32);
        c[j] = __builtin_amdgcn_wmma_f32_16x16x32_bf16(
                 false, __builtin_bit_cast(v16bf, a),
                 false, __builtin_bit_cast(v16bf, b),
                 (short)0, c[j], false, false);
      }
    }
    // Epilogue: relu(c + be) dotted into 4 per-task partial sums (per-lane n column)
    #pragma unroll
    for (int j = 0; j < 4; ++j) {
      const int n = nbase + j * 16 + lmod;
      const float bias = be[e * H_DIM + n];
      const float wf0 = Wf[0 * H_DIM + n], wf1 = Wf[1 * H_DIM + n];
      const float wf2 = Wf[2 * H_DIM + n], wf3 = Wf[3 * H_DIM + n];
      #pragma unroll
      for (int r = 0; r < 8; ++r) {
        float v = fmaxf(c[j][r] + bias, 0.f);
        sacc[0][r] = fmaf(v, wf0, sacc[0][r]);
        sacc[1][r] = fmaf(v, wf1, sacc[1][r]);
        sacc[2][r] = fmaf(v, wf2, sacc[2][r]);
        sacc[3][r] = fmaf(v, wf3, sacc[3][r]);
      }
    }
  }

  // Reduce over the 16 lanes sharing one matrix row, then store s[b, e, t]
  #pragma unroll
  for (int t = 0; t < T_NUM; ++t) {
    #pragma unroll
    for (int r = 0; r < 8; ++r) {
      float v = sacc[t][r];
      v += __shfl_xor(v, 1, 32);
      v += __shfl_xor(v, 2, 32);
      v += __shfl_xor(v, 4, 32);
      v += __shfl_xor(v, 8, 32);
      if (lmod == 0) {
        const int m = w * 16 + lhalf * 8 + r;
        s_out[(size_t)(b0 + m) * (E_NUM * T_NUM) + e * T_NUM + t] = v;
      }
    }
  }
}

// -------- Kernel 3: gate GEMM -> logits[B, T*E] --------
__global__ __launch_bounds__(256, 1) void mmoe_gate_kernel(
    const float* __restrict__ x, const unsigned short* __restrict__ WgB,
    const float* __restrict__ bg, float* __restrict__ logits) {
  extern __shared__ __align__(16) char smem[];
  unsigned short* xs = reinterpret_cast<unsigned short*>(smem);
  const int tid = threadIdx.x;
  const int b0  = blockIdx.x * BM;

  stage_x(xs, x, b0, tid);
  __syncthreads();

  const int lane  = tid & 31;
  const int w     = tid >> 5;
  const int lhalf = lane >> 4;
  const int lmod  = lane & 15;
  const unsigned short* xrow = xs + (w * 16 + lmod) * LDSROW;
  const int kA = lhalf * 16;

  v8f c[4] = {};                                       // 4 N-chunks cover T*E = 64 cols
  #pragma unroll 4
  for (int kc = 0; kc < I_DIM / 32; ++kc) {
    const int k0 = kc * 32 + kA;
    v8u alo = *reinterpret_cast<const v8u*>(xrow + k0);
    v8u ahi = *reinterpret_cast<const v8u*>(xrow + k0 + 8);
    v16u a  = __builtin_shufflevector(alo, ahi, 0,1,2,3,4,5,6,7,8,9,10,11,12,13,14,15);
    #pragma unroll
    for (int nc = 0; nc < 4; ++nc) {
      const unsigned short* brow = WgB + (size_t)(nc * 16 + lmod) * I_DIM + lhalf * 16;
      v16u b = *reinterpret_cast<const v16u*>(brow + kc * 32);
      c[nc] = __builtin_amdgcn_wmma_f32_16x16x32_bf16(
                false, __builtin_bit_cast(v16bf, a),
                false, __builtin_bit_cast(v16bf, b),
                (short)0, c[nc], false, false);
    }
  }
  #pragma unroll
  for (int nc = 0; nc < 4; ++nc) {
    const float bias = bg[nc * 16 + lmod];
    #pragma unroll
    for (int r = 0; r < 8; ++r) {
      const int m = w * 16 + lhalf * 8 + r;
      logits[(size_t)(b0 + m) * (T_NUM * E_NUM) + nc * 16 + lmod] = c[nc][r] + bias;
    }
  }
}

// -------- Kernel 4: softmax over experts + gated combine + bias --------
__global__ void mmoe_combine_kernel(const float* __restrict__ logits,
                                    const float* __restrict__ s,
                                    const float* __restrict__ bfv,
                                    float* __restrict__ out) {
  const int b = blockIdx.x * blockDim.x + threadIdx.x;
  const float* lb = logits + (size_t)b * (T_NUM * E_NUM);
  const float* sb = s + (size_t)b * (E_NUM * T_NUM);
  #pragma unroll
  for (int t = 0; t < T_NUM; ++t) {
    float l[E_NUM];
    float mx = -3.402823466e38f;
    #pragma unroll
    for (int e = 0; e < E_NUM; ++e) { l[e] = lb[t * E_NUM + e]; mx = fmaxf(mx, l[e]); }
    float den = 0.f, acc = 0.f;
    #pragma unroll
    for (int e = 0; e < E_NUM; ++e) {
      const float p = __expf(l[e] - mx);
      den += p;
      acc = fmaf(p, sb[e * T_NUM + t], acc);
    }
    out[t * B_TOK + b] = acc / den + bfv[t];
  }
}

extern "C" void kernel_launch(void* const* d_in, const int* in_sizes, int n_in,
                              void* d_out, int out_size, void* d_ws, size_t ws_size,
                              hipStream_t stream) {
  (void)in_sizes; (void)n_in; (void)out_size; (void)ws_size;
  const float* x   = (const float*)d_in[0];
  const float* We  = (const float*)d_in[1];
  const float* be  = (const float*)d_in[2];
  const float* Wg  = (const float*)d_in[3];
  const float* bg  = (const float*)d_in[4];
  const float* Wf  = (const float*)d_in[5];
  const float* bfv = (const float*)d_in[6];
  float* out = (float*)d_out;

  char* ws = (char*)d_ws;                               // layout (bytes):
  unsigned short* WeB = (unsigned short*)(ws);                           // 16 MB
  unsigned short* WgB = (unsigned short*)(ws + (size_t)16777216);        // 128 KB
  float* s_buf  = (float*)(ws + (size_t)16777216 + 131072);              // 2 MB
  float* lg_buf = (float*)(ws + (size_t)16777216 + 131072 + 2097152);    // 2 MB

  (void)hipFuncSetAttribute(reinterpret_cast<const void*>(mmoe_expert_kernel),
                            hipFuncAttributeMaxDynamicSharedMemorySize, LDS_BYTES);
  (void)hipFuncSetAttribute(reinterpret_cast<const void*>(mmoe_gate_kernel),
                            hipFuncAttributeMaxDynamicSharedMemorySize, LDS_BYTES);

  mmoe_convert_kernel<<<8256, 256, 0, stream>>>(We, Wg, WeB, WgB);
  mmoe_expert_kernel<<<dim3(B_TOK / BM, E_NUM), 256, LDS_BYTES, stream>>>(x, WeB, be, Wf, s_buf);
  mmoe_gate_kernel<<<B_TOK / BM, 256, LDS_BYTES, stream>>>(x, WgB, bg, lg_buf);
  mmoe_combine_kernel<<<B_TOK / 256, 256, 0, stream>>>(lg_buf, s_buf, bfv, out);
}
